// BasicAttn_1271310319936
// MI455X (gfx1250) — matrix operand order repaired and checked by reading, not compiled, not run on hardware
//
#include <hip/hip_runtime.h>

// ---------------------------------------------------------------------------
// Problem constants: values [Tv,B,V] f32, keys [B,Tk,K] f32, w [K,V] f32
// ---------------------------------------------------------------------------
constexpr int TV = 2048;
constexpr int NB = 16;    // batch
constexpr int VD = 1024;  // V
constexpr int TK = 2048;
constexpr int KD = 1024;  // K

// Tiling: block 128x128, 8 waves (4 in M x 2 in N), wave tile 32x64 -> 8 WMMA/step
constexpr int BM = 128;
constexpr int BN = 128;
constexpr int BK = 32;
constexpr int LDT = 40;   // LDS row stride in shorts (80B, keeps 16B alignment)

typedef __attribute__((ext_vector_type(16))) __bf16 bf16x16;
typedef __attribute__((ext_vector_type(8)))  float  v8f;
typedef __attribute__((ext_vector_type(8)))  unsigned short u16x8;

struct Frag32B { u16x8 lo, hi; };

static __device__ __forceinline__ unsigned short cvt_bf16(float f) {
    __bf16 h = (__bf16)f;                       // hardware RNE f32->bf16 (v_cvt_pk_bf16_f32)
    return __builtin_bit_cast(unsigned short, h);
}

// A fragment: 16x32 bf16, lane holds row m=lane&15.
// elems 0..7 = K[hl*8..+7], elems 8..15 = K[16+hl*8..+7]
static __device__ __forceinline__ bf16x16 load_frag_a(const unsigned short* rowp, int hl) {
    Frag32B f;
    f.lo = *(const u16x8*)(rowp + hl * 8);
    f.hi = *(const u16x8*)(rowp + 16 + hl * 8);
    return __builtin_bit_cast(bf16x16, f);
}

// B fragment: B^T stored [n][k] in LDS; lane holds col n=lane&15, K[hl*8..+7], K[16+hl*8..+7]
static __device__ __forceinline__ bf16x16 load_frag_b(const unsigned short* rowp, int hl) {
    Frag32B f;
    f.lo = *(const u16x8*)(rowp + hl * 8);
    f.hi = *(const u16x8*)(rowp + 16 + hl * 8);
    return __builtin_bit_cast(bf16x16, f);
}

// 32x64 wave tile: 2 A frags, 4 B frags, 8 WMMAs
static __device__ __forceinline__ void mma_tiles(const unsigned short* sA,
                                                 const unsigned short* sB,
                                                 v8f acc[2][4], int wm, int wn, int lane) {
    const int r  = lane & 15;
    const int hl = lane >> 4;
    bf16x16 a[2], bb[4];
    a[0] = load_frag_a(sA + (wm * 32 +      r) * LDT, hl);
    a[1] = load_frag_a(sA + (wm * 32 + 16 + r) * LDT, hl);
#pragma unroll
    for (int j = 0; j < 4; ++j)
        bb[j] = load_frag_b(sB + (wn * 64 + j * 16 + r) * LDT, hl);
#pragma unroll
    for (int i = 0; i < 2; ++i)
#pragma unroll
        for (int j = 0; j < 4; ++j)
            acc[i][j] = __builtin_amdgcn_wmma_f32_16x16x32_bf16(
                false, a[i], false, bb[j], (short)0, acc[i][j], false, false);
}

static __device__ __forceinline__ void zero_acc(v8f acc[2][4]) {
    const v8f z = {0.f, 0.f, 0.f, 0.f, 0.f, 0.f, 0.f, 0.f};
#pragma unroll
    for (int i = 0; i < 2; ++i)
#pragma unroll
        for (int j = 0; j < 4; ++j) acc[i][j] = z;
}

// Async global->LDS 32B copy (2 x b128), gfx1250 ASYNCcnt path.
static __device__ __forceinline__ void async_copy32(unsigned short* lds_dst,
                                                    const unsigned short* gsrc) {
    const unsigned lds_a = (unsigned)(uintptr_t)(void*)lds_dst;    // low 32b = LDS offset
    const unsigned long long ga = (unsigned long long)(uintptr_t)gsrc;
    asm volatile("global_load_async_to_lds_b128 %0, %1, off"
                 :: "v"(lds_a), "v"(ga) : "memory");
    asm volatile("global_load_async_to_lds_b128 %0, %1, off offset:16"
                 :: "v"(lds_a), "v"(ga) : "memory");
}
static __device__ __forceinline__ void async_wait0() {
    asm volatile("s_wait_asynccnt 0x0" ::: "memory");
}

// ---------------------------------------------------------------------------
// values f32 -> bf16 pre-conversion (done once; removes per-tile re-conversion)
// ---------------------------------------------------------------------------
__global__ __launch_bounds__(256) void k_cvt(const float* __restrict__ src,
                                             unsigned short* __restrict__ dst) {
    const size_t base = ((size_t)blockIdx.x * 256 + threadIdx.x) * 8;
#pragma unroll
    for (int i = 0; i < 8; ++i) dst[base + i] = cvt_bf16(src[base + i]);
}

// ---------------------------------------------------------------------------
// GEMM1: part[b] (bf16) = keys[b] (2048x1024 f32) @ w (1024x1024 f32)
// Double-buffered LDS: fills for step k+1 overlap WMMAs of step k.
// ---------------------------------------------------------------------------
__global__ __launch_bounds__(256) void k_proj(const float* __restrict__ keys,
                                              const float* __restrict__ w,
                                              unsigned short* __restrict__ part) {
    __shared__ __align__(16) unsigned short sA[2][BM * LDT];
    __shared__ __align__(16) unsigned short sB[2][BN * LDT];
    const int b  = blockIdx.z;
    const int m0 = blockIdx.y * BM;
    const int n0 = blockIdx.x * BN;
    const int tid = threadIdx.x, lane = tid & 31, wave = tid >> 5;
    const int wm = wave & 3, wn = wave >> 2;
    const float* A = keys + (size_t)b * TK * KD;

    auto fillA = [&](unsigned short* dst_t, int k0) {
        const int r = tid >> 1, kh = (tid & 1) * 16;
        const float* src = A + (size_t)(m0 + r) * KD + k0 + kh;
        unsigned short* dst = dst_t + r * LDT + kh;
        if (k0 + BK < KD) __builtin_prefetch(src + BK, 0, 0);
#pragma unroll
        for (int i = 0; i < 16; ++i) dst[i] = cvt_bf16(src[i]);
    };
    auto fillB = [&](unsigned short* dst_t, int k0) {
#pragma unroll
        for (int j = 0; j < 16; ++j) {   // sB[n][k] = w[k0+k][n0+n]
            const int idx = tid + 256 * j;
            const int k = idx >> 7, n = idx & 127;
            dst_t[n * LDT + k] = cvt_bf16(w[(size_t)(k0 + k) * VD + n0 + n]);
        }
    };

    v8f acc[2][4];
    zero_acc(acc);

    fillA(sA[0], 0);
    fillB(sB[0], 0);
    int cur = 0;
    for (int k0 = 0; k0 < KD; k0 += BK) {
        __syncthreads();                       // fills of cur visible block-wide
        if (k0 + BK < KD) { fillA(sA[cur ^ 1], k0 + BK); fillB(sB[cur ^ 1], k0 + BK); }
        mma_tiles(sA[cur], sB[cur], acc, wm, wn, lane);
        __syncthreads();                       // readers of cur done before reuse
        cur ^= 1;
    }

    const int r = lane & 15, hl = lane >> 4;
#pragma unroll
    for (int i = 0; i < 2; ++i)
#pragma unroll
        for (int j = 0; j < 4; ++j)
#pragma unroll
            for (int e = 0; e < 8; ++e) {
                const int m = m0 + wm * 32 + i * 16 + hl * 8 + e;
                const int n = n0 + wn * 64 + j * 16 + r;
                part[(size_t)b * TK * VD + (size_t)m * VD + n] = cvt_bf16(acc[i][j][e]);
            }
}

// ---------------------------------------------------------------------------
// GEMM2: logits[b] = part[b] (2048x1024 bf16) @ values_t[b] (1024x2048)
//        Both tiles are bf16 copies -> async DMA to LDS, double-buffered.
// ---------------------------------------------------------------------------
template <bool PRE>
__global__ __launch_bounds__(256) void k_logits(const unsigned short* __restrict__ part,
                                                const float* __restrict__ values,
                                                const unsigned short* __restrict__ values_bf,
                                                float* __restrict__ attn) {
    __shared__ __align__(16) unsigned short sA[2][BM * LDT];
    __shared__ __align__(16) unsigned short sB[2][BN * LDT];
    const int b  = blockIdx.z;
    const int m0 = blockIdx.y * BM;
    const int s0 = blockIdx.x * BN;
    const int tid = threadIdx.x, lane = tid & 31, wave = tid >> 5;
    const int wm = wave & 3, wn = wave >> 2;
    const unsigned short* A = part + (size_t)b * TK * VD;

    auto fill = [&](unsigned short* dstA, unsigned short* dstB, int k0) {
        const int r = tid >> 1, kh = (tid & 1) * 16;
        { // A tile: bf16 ws -> LDS via async DMA
            const unsigned short* src = A + (size_t)(m0 + r) * VD + k0 + kh;
            if (k0 + BK < VD) __builtin_prefetch(src + BK, 0, 0);
            async_copy32(dstA + r * LDT + kh, src);
        }
        if constexpr (PRE) { // B^T tile: bf16 copy via async DMA
            const unsigned short* src =
                values_bf + (size_t)(s0 + r) * NB * VD + (size_t)b * VD + k0 + kh;
            async_copy32(dstB + r * LDT + kh, src);
        } else {             // B^T tile: convert f32 -> bf16
            const float* src = values + (size_t)(s0 + r) * NB * VD + (size_t)b * VD + k0 + kh;
            unsigned short* dst = dstB + r * LDT + kh;
#pragma unroll
            for (int i = 0; i < 16; ++i) dst[i] = cvt_bf16(src[i]);
        }
    };

    v8f acc[2][4];
    zero_acc(acc);

    fill(sA[0], sB[0], 0);
    int cur = 0;
    for (int k0 = 0; k0 < VD; k0 += BK) {
        async_wait0();                         // my fills of cur have landed in LDS
        __syncthreads();                       // ... and everyone else's too
        if (k0 + BK < VD) fill(sA[cur ^ 1], sB[cur ^ 1], k0 + BK);  // DMA under the mma
        mma_tiles(sA[cur], sB[cur], acc, wm, wn, lane);
        __syncthreads();                       // readers of cur done before reuse
        cur ^= 1;
    }

    const int r = lane & 15, hl = lane >> 4;
#pragma unroll
    for (int i = 0; i < 2; ++i)
#pragma unroll
        for (int j = 0; j < 4; ++j)
#pragma unroll
            for (int e = 0; e < 8; ++e) {
                const int m = m0 + wm * 32 + i * 16 + hl * 8 + e;
                const int s = s0 + wn * 64 + j * 16 + r;
                attn[(size_t)b * TK * TV + (size_t)m * TV + s] = acc[i][j][e];
            }
}

// ---------------------------------------------------------------------------
// Row softmax over Tv=2048, in place. One 256-thread block per (b,t) row.
// ---------------------------------------------------------------------------
__global__ __launch_bounds__(256) void k_softmax(float* __restrict__ attn) {
    __shared__ float red[256];
    float* row = attn + (size_t)blockIdx.x * TV;
    const int tid = threadIdx.x;

    float mx = -__builtin_inff();
#pragma unroll
    for (int i = 0; i < 8; ++i) mx = fmaxf(mx, row[tid + 256 * i]);
    red[tid] = mx;
    __syncthreads();
    for (int s = 128; s > 0; s >>= 1) {
        if (tid < s) red[tid] = fmaxf(red[tid], red[tid + s]);
        __syncthreads();
    }
    mx = red[0];
    __syncthreads();

    float ev[8];
    float sum = 0.f;
#pragma unroll
    for (int i = 0; i < 8; ++i) {
        ev[i] = __expf(row[tid + 256 * i] - mx);
        sum += ev[i];
    }
    red[tid] = sum;
    __syncthreads();
    for (int s = 128; s > 0; s >>= 1) {
        if (tid < s) red[tid] += red[tid + s];
        __syncthreads();
    }
    const float inv = 1.0f / red[0];
#pragma unroll
    for (int i = 0; i < 8; ++i) row[tid + 256 * i] = ev[i] * inv;
}

// ---------------------------------------------------------------------------
// GEMM3: out[b] = attn[b] (2048x2048 f32) @ values[:,b,:] (2048x1024)
// ---------------------------------------------------------------------------
template <bool PRE>
__global__ __launch_bounds__(256) void k_out(const float* __restrict__ attn,
                                             const float* __restrict__ values,
                                             const unsigned short* __restrict__ values_bf,
                                             float* __restrict__ out) {
    __shared__ __align__(16) unsigned short sA[2][BM * LDT];
    __shared__ __align__(16) unsigned short sB[2][BN * LDT];
    const int b  = blockIdx.z;
    const int m0 = blockIdx.y * BM;
    const int n0 = blockIdx.x * BN;
    const int tid = threadIdx.x, lane = tid & 31, wave = tid >> 5;
    const int wm = wave & 3, wn = wave >> 2;
    const float* A = attn + (size_t)b * TK * TV;

    auto fillA = [&](unsigned short* dst_t, int k0) {
        const int r = tid >> 1, kh = (tid & 1) * 16;
        const float* src = A + (size_t)(m0 + r) * TV + k0 + kh;
        unsigned short* dst = dst_t + r * LDT + kh;
        if (k0 + BK < TV) __builtin_prefetch(src + BK, 0, 0);
#pragma unroll
        for (int i = 0; i < 16; ++i) dst[i] = cvt_bf16(src[i]);
    };
    auto fillB = [&](unsigned short* dst_t, int k0) {
#pragma unroll
        for (int j = 0; j < 16; ++j) {   // sB[n][k] = values[k0+k][b][n0+n]
            const int idx = tid + 256 * j;
            const int k = idx >> 7, n = idx & 127;
            const size_t off = (size_t)(k0 + k) * NB * VD + (size_t)b * VD + n0 + n;
            dst_t[n * LDT + k] = PRE ? values_bf[off] : cvt_bf16(values[off]);
        }
    };

    v8f acc[2][4];
    zero_acc(acc);

    fillA(sA[0], 0);
    fillB(sB[0], 0);
    int cur = 0;
    for (int k0 = 0; k0 < TV; k0 += BK) {
        __syncthreads();
        if (k0 + BK < TV) { fillA(sA[cur ^ 1], k0 + BK); fillB(sB[cur ^ 1], k0 + BK); }
        mma_tiles(sA[cur], sB[cur], acc, wm, wn, lane);
        __syncthreads();
        cur ^= 1;
    }

    const int r = lane & 15, hl = lane >> 4;
#pragma unroll
    for (int i = 0; i < 2; ++i)
#pragma unroll
        for (int j = 0; j < 4; ++j)
#pragma unroll
            for (int e = 0; e < 8; ++e) {
                const int m = m0 + wm * 32 + i * 16 + hl * 8 + e;
                const int n = n0 + wn * 64 + j * 16 + r;
                out[(size_t)b * TK * VD + (size_t)m * VD + n] = acc[i][j][e];
            }
}

// ---------------------------------------------------------------------------
// Launcher. Inputs: [0]=values f32, [1]=values_mask i32 (unused), [2]=keys f32,
//                   [3]=w f32. d_out = attn_dist (B*Tk*Tv) ++ output (B*Tk*V).
// d_ws: [part bf16: 64MB][values bf16: 64MB if ws_size allows]
// ---------------------------------------------------------------------------
extern "C" void kernel_launch(void* const* d_in, const int* in_sizes, int n_in,
                              void* d_out, int out_size, void* d_ws, size_t ws_size,
                              hipStream_t stream) {
    (void)in_sizes; (void)n_in; (void)out_size;
    const float* values = (const float*)d_in[0];
    const float* keys   = (const float*)d_in[2];
    const float* w      = (const float*)d_in[3];

    float* attn = (float*)d_out;                         // [B, Tk, Tv]
    float* out  = (float*)d_out + (size_t)NB * TK * TV;  // [B, Tk, V]

    unsigned short* part = (unsigned short*)d_ws;                  // [B, Tk, V] bf16
    unsigned short* vbf  = part + (size_t)NB * TK * VD;            // [Tv, B, V] bf16
    const size_t need_pre = ((size_t)NB * TK * VD + (size_t)TV * NB * VD) * 2;
    const bool pre = ws_size >= need_pre;

    dim3 blk(256);
    if (pre)
        k_cvt<<<dim3((TV * NB * VD) / (256 * 8)), blk, 0, stream>>>(values, vbf);

    k_proj<<<dim3(VD / BN, TK / BM, NB), blk, 0, stream>>>(keys, w, part);

    if (pre)
        k_logits<true><<<dim3(TV / BN, TK / BM, NB), blk, 0, stream>>>(part, values, vbf, attn);
    else
        k_logits<false><<<dim3(TV / BN, TK / BM, NB), blk, 0, stream>>>(part, values, vbf, attn);

    k_softmax<<<dim3(NB * TK), blk, 0, stream>>>(attn);

    if (pre)
        k_out<true><<<dim3(VD / BN, TK / BM, NB), blk, 0, stream>>>(attn, values, vbf, out);
    else
        k_out<false><<<dim3(VD / BN, TK / BM, NB), blk, 0, stream>>>(attn, values, vbf, out);
}
